// AnsiToPixelsLayer_51599737094676
// MI455X (gfx1250) — matrix operand order repaired and checked by reading, not compiled, not run on hardware
//
#include <hip/hip_runtime.h>
#include <stdint.h>

typedef _Float16 v16h __attribute__((ext_vector_type(16)));
typedef _Float16 v8h  __attribute__((ext_vector_type(8)));
typedef float    v8f  __attribute__((ext_vector_type(8)));
typedef float    v4f  __attribute__((ext_vector_type(4)));

#define GRID_HH   20
#define GRID_WW   80
#define NCH       264        // 256 glyph weights + 8 color channels
#define NPIX      128        // 16x8 glyph pixels = N dimension
#define M_TILES   12800      // 128*20*80 cells / 16
#define ASTRIDE   264        // padded row stride (halfs) for Ahi/Alo: 2-way max conflicts

// Dynamic LDS (131072 bytes total), two temporally-disjoint layouts:
//  Phase 0 (B init):  [0,65536) Bhi frags, [65536,131072) Blo frags
//  Main loop (after B is register-resident):
//    [0,     8448)  Ahi : 16 rows x 264 halfs (256 used + 8 pad)
//    [8448, 16896)  Alo
//    [16896,17408)  Col : 16 x 8 f32 (fg_bold,fg_rgb,bg_bold,bg_rgb)
//    [17408,34304)  Raw0: 16 x 264 f32 (async-copy target)
//    [34304,51200)  Raw1
#define SMEM_BYTES 131072
#define OFF_ALO    8448
#define OFF_COL    16896
#define OFF_RAW0   17408
#define OFF_RAW1   34304

static __device__ __forceinline__ v16h cat16(v8h a, v8h b) {
  v16h r;
#pragma unroll
  for (int j = 0; j < 8; ++j) { r[j] = a[j]; r[j + 8] = b[j]; }
  return r;
}

__global__ __launch_bounds__(256)
void ansi_to_pixels_wmma(const float* __restrict__ data,
                         const float* __restrict__ cm,
                         float* __restrict__ out,
                         int numTiles) {
  extern __shared__ char smem[];
  const int tid  = threadIdx.x;
  const int lane = tid & 31;
  const int wv   = tid >> 5;          // wave id == N-tile id (0..7)

  // ---- Phase 0: split char_matrix into f16 hi/lo WMMA B-fragments in LDS.
  // B 32x16 frags, this code's contiguous k order: lane L -> N = L&15; lanes 0-15 hold K 0..15
  // (element j = K), lanes 16-31 hold K 16..31 (element j = K-16).
  {
    _Float16* Bhi = (_Float16*)smem;
    _Float16* Blo = (_Float16*)(smem + 65536);
    for (int i = tid; i < 8 * 8 * 32 * 16; i += 256) {
      int j  = i & 15;
      int ln = (i >> 4) & 31;
      int nt = (i >> 9) & 7;
      int kt = i >> 12;
      int n  = nt * 16 + (ln & 15);
      int kk = (ln < 16) ? j : (j + 16);
      float v = cm[(kt * 32 + kk) * NPIX + n];
      _Float16 h = (_Float16)v;
      Bhi[i] = h;
      Blo[i] = (_Float16)(v - (float)h);
    }
  }
  __syncthreads();

  // Each wave keeps its N-tile's B fragments in registers for all K-tiles.
  v16h bh[8], bl[8];
#pragma unroll
  for (int kt = 0; kt < 8; ++kt) {
    const _Float16* p = (const _Float16*)smem + ((kt * 8 + wv) * 32 + lane) * 16;
    const _Float16* q = (const _Float16*)(smem + 65536) + ((kt * 8 + wv) * 32 + lane) * 16;
    bh[kt] = cat16(*(const v8h*)p, *(const v8h*)(p + 8));
    bl[kt] = cat16(*(const v8h*)q, *(const v8h*)(q + 8));
  }
  __syncthreads();   // B reads done; LDS region is now reusable

  _Float16* Ahi = (_Float16*)smem;
  _Float16* Alo = (_Float16*)(smem + OFF_ALO);
  float*    Col = (float*)(smem + OFF_COL);
  float*    Raw0 = (float*)(smem + OFF_RAW0);
  float*    Raw1 = (float*)(smem + OFF_RAW1);

  // Async copy of one 16-cell tile (16x264 f32, 1056 float4s) into Raw[buf].
  auto prefetch_tile = [&](int tile, float* rawbuf) {
    const float* src = data + (long)tile * 16 * NCH;
    unsigned ldsbase = (unsigned)(uintptr_t)(void*)rawbuf;
    for (int i = tid; i < 16 * 66; i += 256) {
      unsigned     ldsa = ldsbase + (unsigned)i * 16u;
      uint64_t     ga   = (uint64_t)(uintptr_t)(src + i * 4);
      asm volatile("global_load_async_to_lds_b128 %0, %1, off"
                   :: "v"(ldsa), "v"(ga) : "memory");
    }
  };

  if (blockIdx.x < (unsigned)numTiles) prefetch_tile(blockIdx.x, Raw0);

  int buf = 0;
  for (int tile = blockIdx.x; tile < numTiles; tile += gridDim.x, buf ^= 1) {
    float* RawC = buf ? Raw1 : Raw0;   // current tile's raw data
    float* RawN = buf ? Raw0 : Raw1;   // next tile's target

    asm volatile("s_wait_asynccnt 0x0" ::: "memory");
    __syncthreads();   // Raw[buf] landed; previous Ahi/Alo readers finished

    int nextTile = tile + gridDim.x;
    if (nextTile < numTiles) prefetch_tile(nextTile, RawN);

    // ---- Cooperative one-shot f32 -> f16 hi/lo conversion (no redundancy).
    {
      int row = tid >> 4;
      int k0  = (tid & 15) << 4;      // 16 consecutive k per thread
      const float* rp = RawC + row * NCH + k0;
      v4f f0 = *(const v4f*)(rp);
      v4f f1 = *(const v4f*)(rp + 4);
      v4f f2 = *(const v4f*)(rp + 8);
      v4f f3 = *(const v4f*)(rp + 12);
      float vals[16] = {f0.x, f0.y, f0.z, f0.w, f1.x, f1.y, f1.z, f1.w,
                        f2.x, f2.y, f2.z, f2.w, f3.x, f3.y, f3.z, f3.w};
      v8h h0, h1, l0, l1;
#pragma unroll
      for (int j = 0; j < 8; ++j) {
        _Float16 a = (_Float16)vals[j];
        _Float16 b = (_Float16)vals[j + 8];
        h0[j] = a; l0[j] = (_Float16)(vals[j] - (float)a);
        h1[j] = b; l1[j] = (_Float16)(vals[j + 8] - (float)b);
      }
      _Float16* dh = Ahi + row * ASTRIDE + k0;
      _Float16* dl = Alo + row * ASTRIDE + k0;
      *(v8h*)dh = h0; *(v8h*)(dh + 8) = h1;
      *(v8h*)dl = l0; *(v8h*)(dl + 8) = l1;
      if (tid < 16) {   // colors straight through as f32
        const float* cp = RawC + tid * NCH + 256;
        *(v4f*)(Col + tid * 8)     = *(const v4f*)cp;
        *(v4f*)(Col + tid * 8 + 4) = *(const v4f*)(cp + 4);
      }
    }
    __syncthreads();

    // ---- GEMM: 16 cells x 16 pixels (this wave's N-tile), K=256 in 8 steps.
    // A frags, same contiguous k order as B: lanes 0-15: K 0..15; lanes 16-31: K 16..31.
    v8f acc = {};
    const int koff = (lane < 16) ? 0 : 16;
    const int row  = lane & 15;
#pragma unroll
    for (int kt = 0; kt < 8; ++kt) {
      const _Float16* ah = Ahi + row * ASTRIDE + kt * 32 + koff;
      const _Float16* al = Alo + row * ASTRIDE + kt * 32 + koff;
      v16h afh = cat16(*(const v8h*)ah, *(const v8h*)(ah + 8));
      v16h afl = cat16(*(const v8h*)al, *(const v8h*)(al + 8));
      acc = __builtin_amdgcn_wmma_f32_16x16x32_f16(false, afh, false, bh[kt],
                                                   (short)0, acc, false, false);
      acc = __builtin_amdgcn_wmma_f32_16x16x32_f16(false, afl, false, bh[kt],
                                                   (short)0, acc, false, false);
      acc = __builtin_amdgcn_wmma_f32_16x16x32_f16(false, afh, false, bl[kt],
                                                   (short)0, acc, false, false);
    }

    // ---- Epilogue: blend with fg/bg, scatter to pixel image.
    // C/D layout: vgpr r, lanes 0-15: (M=r, N=lane); lanes 16-31: (M=r+8, N-16)
    int m0  = tile * 16;
    int b0  = m0 / (GRID_HH * GRID_WW);
    int rem = m0 % (GRID_HH * GRID_WW);
    int h0  = rem / GRID_WW;
    int w0  = rem % GRID_WW;          // 16-cell tile never crosses a grid row
    int p   = wv * 16 + (lane & 15);  // glyph pixel index
    int y   = p >> 3, x = p & 7;
    int rowBase = (b0 * 320 + h0 * 16 + y) * 640 * 3;
    int mhi = (lane >> 4) << 3;
#pragma unroll
    for (int r = 0; r < 8; ++r) {
      int M = r + mhi;
      v4f fq = *(const v4f*)(Col + M * 8);      // fg_bold, fg_rgb
      v4f gq = *(const v4f*)(Col + M * 8 + 4);  // bg_bold, bg_rgb
      float fb = 0.5f * fq.x + 0.5f;
      float gb = 0.5f * gq.x + 0.5f;
      float fr = fb * fq.y, fgc = fb * fq.z, fbc = fb * fq.w;
      float gr = gb * gq.y, ggc = gb * gq.z, gbc = gb * gq.w;
      float raw = acc[r];
      int idx = rowBase + ((w0 + M) * 8 + x) * 3;
      out[idx + 0] = gr  + raw * (fr  - gr);
      out[idx + 1] = ggc + raw * (fgc - ggc);
      out[idx + 2] = gbc + raw * (fbc - gbc);
    }
  }
}

extern "C" void kernel_launch(void* const* d_in, const int* in_sizes, int n_in,
                              void* d_out, int out_size, void* d_ws, size_t ws_size,
                              hipStream_t stream) {
  const float* data = (const float*)d_in[0];   // (128,20,80,264) f32
  const float* cm   = (const float*)d_in[1];   // (256,16,8) f32
  float* out        = (float*)d_out;           // (128,320,640,3) f32
  (void)in_sizes; (void)n_in; (void)out_size; (void)d_ws; (void)ws_size;

  dim3 grid(1280), block(256);
  ansi_to_pixels_wmma<<<grid, block, SMEM_BYTES, stream>>>(data, cm, out, M_TILES);
}
